// Net_GAT_3994319585978
// MI455X (gfx1250) — compile-verified, matrix operand order checked by reading, not run on hardware
//
#include <hip/hip_runtime.h>
#include <hip/hip_bf16.h>

// ---------------------------------------------------------------------------
// Net_GAT pipeline for MI455X (gfx1250, wave32, WMMA).
// Sizes (hard-coded from reference): B=16 T=256 G=1024 P=16 N=4096 GO=6
// H=2 C=4 E=65536 NUM_OUT=32
// ---------------------------------------------------------------------------

#define BATCH   16
#define T_TF    256
#define G_GENES 1024
#define P_PK    16
#define N_GO    4096
#define GO_DIM  6
#define NFC1    (N_GO * GO_DIM)     // 24576
#define HEADS   2
#define OUTC    4
#define NEDGE   65536
#define NUM_OUT 32
#define KREAD   (N_GO * OUTC)       // 16384

typedef __attribute__((ext_vector_type(16))) __bf16 v16bf;
typedef __attribute__((ext_vector_type(8)))  float  v8f;
typedef __attribute__((ext_vector_type(4)))  int    v4i;

#if defined(__has_builtin)
#if __has_builtin(__builtin_amdgcn_global_load_async_to_lds_b128) && \
    __has_builtin(__builtin_amdgcn_s_wait_asynccnt)
#define USE_ASYNC_LDS 1
#endif
#endif

// address-space-qualified b128 pointers for the async-to-LDS builtin
typedef __attribute__((address_space(1))) v4i gv4i;  // global
typedef __attribute__((address_space(3))) v4i lv4i;  // LDS (32-bit ptr)

__device__ __forceinline__ __bf16 tob(float f) { return (__bf16)f; }  // RNE cvt
__device__ __forceinline__ float lrelu(float v, float s) {
  return v >= 0.f ? v : s * v;
}

// ---------------------------------------------------------------------------
// Stage 1: per-gene subnets. x_cat[b,g] = lrelu(dot(x[b,:,g,:], W_sub[g]) + b_sub[g])
// One block per gene; W_sub[g] staged into LDS (async-to-LDS on gfx1250);
// lanes map to p for 64B coalesced reads of the 256 MB x stream.
// ---------------------------------------------------------------------------
__global__ void subnet_kernel(const float* __restrict__ x,
                              const float* __restrict__ W_sub,
                              const float* __restrict__ b_sub,
                              float* __restrict__ x_cat) {
  __shared__ float Wl[T_TF * P_PK];   // 16 KB
  __shared__ float red[256];
  const int g   = blockIdx.x;
  const int tid = threadIdx.x;

#ifdef USE_ASYNC_LDS
  {
    const float* gsrc = W_sub + (size_t)g * (T_TF * P_PK);
    for (int i = tid * 4; i < T_TF * P_PK; i += 256 * 4) {
      __builtin_amdgcn_global_load_async_to_lds_b128(
          (gv4i*)(uintptr_t)(gsrc + i),
          (lv4i*)(unsigned)(uintptr_t)(Wl + i), 0, 0);
    }
    __builtin_amdgcn_s_wait_asynccnt(0);
  }
#else
  for (int i = tid; i < T_TF * P_PK; i += 256)
    Wl[i] = W_sub[(size_t)g * (T_TF * P_PK) + i];
#endif
  __syncthreads();

  const int p  = tid & 15;
  const int t0 = tid >> 4;
  for (int b = 0; b < BATCH; ++b) {
    float acc = 0.f;
    const float* xb = x + (size_t)b * T_TF * G_GENES * P_PK;
    for (int t = t0; t < T_TF; t += 16) {
      acc += xb[((size_t)t * G_GENES + g) * P_PK + p] * Wl[t * P_PK + p];
    }
    red[tid] = acc;
    __syncthreads();
    for (int s2 = 128; s2 > 0; s2 >>= 1) {
      if (tid < s2) red[tid] += red[tid + s2];
      __syncthreads();
    }
    if (tid == 0) x_cat[b * G_GENES + g] = lrelu(red[0] + b_sub[g], 0.01f);
    __syncthreads();
  }
}

// ---------------------------------------------------------------------------
// Stage 2: h = lrelu(x_cat @ (fc1_w*mask)^T + fc1_b), stored as [n_go, b, d].
// GEMM M=16 K=1024 N=24576 via v_wmma_f32_16x16x32_bf16.
// One wave per 16-column tile, 8 waves per block. 32 WMMAs per tile.
// ---------------------------------------------------------------------------
__global__ void fc1_wmma_kernel(const float* __restrict__ x_cat,
                                const float* __restrict__ fc1_w,
                                const float* __restrict__ mask_rep,
                                const float* __restrict__ fc1_b,
                                float* __restrict__ h_nbd) {
  const int tile = blockIdx.x * 8 + (threadIdx.x >> 5);   // 0..1535
  const int lane = threadIdx.x & 31;
  const int n0   = tile * 16;
  const int mrow = lane & 15;   // A: M-row (batch) ; B: N-col (fc1 row)
  const int half = lane >> 4;   // K-subset selector per ISA layout

  const float* arow  = x_cat    + (size_t)mrow * G_GENES;
  const float* wrow  = fc1_w    + (size_t)(n0 + mrow) * G_GENES;
  const float* mrowp = mask_rep + (size_t)(n0 + mrow) * G_GENES;

  v8f acc = {};
  for (int kc = 0; kc < G_GENES; kc += 32) {
    if (kc + 128 < G_GENES) {
      __builtin_prefetch(wrow + kc + 128, 0, 3);
      __builtin_prefetch(mrowp + kc + 128, 0, 3);
    }
    const int k0 = kc + 8 * half;        // K = {0..7} or {8..15}
    const int k1 = kc + 16 + 8 * half;   // K = {16..23} or {24..31}
    v16bf A, Bm;
#pragma unroll
    for (int i = 0; i < 8; ++i) {
      A[i]      = tob(arow[k0 + i]);
      A[8 + i]  = tob(arow[k1 + i]);
      Bm[i]     = tob(wrow[k0 + i] * mrowp[k0 + i]);
      Bm[8 + i] = tob(wrow[k1 + i] * mrowp[k1 + i]);
    }
    acc = __builtin_amdgcn_wmma_f32_16x16x32_bf16(
        false, A, false, Bm, (short)0, acc, false, false);
  }

  // D layout: lane -> N col (lane&15), VGPR r -> M row r (+8 for lanes>=16).
  const int   n    = n0 + mrow;
  const int   d    = n >> 12;          // n / 4096
  const int   n_go = n & (N_GO - 1);   // n % 4096
  const float bias = fc1_b[n];
#pragma unroll
  for (int r = 0; r < 8; ++r) {
    const int b = r + 8 * half;
    float v = acc[r] + bias;
    h_nbd[((size_t)n_go * BATCH + b) * GO_DIM + d] = lrelu(v, 0.01f);
  }
}

// ---------------------------------------------------------------------------
// Stage 3: hn = h @ gat_weight  ([6]->[2,4]); precompute per-node attention
// halves: si = <hn[h,:], att_i[h]>, sj = <hn[h,:], att_j[h]>.
// ---------------------------------------------------------------------------
__global__ void gat_proj_kernel(const float* __restrict__ h_nbd,
                                const float* __restrict__ gat_w,
                                const float* __restrict__ gat_att,
                                float* __restrict__ hn,
                                float* __restrict__ si,
                                float* __restrict__ sj) {
  const int idx = blockIdx.x * blockDim.x + threadIdx.x;  // n*16 + b
  if (idx >= N_GO * BATCH) return;
  const float* hv = h_nbd + (size_t)idx * GO_DIM;
  float h6[GO_DIM];
#pragma unroll
  for (int d = 0; d < GO_DIM; ++d) h6[d] = hv[d];
  float o8[HEADS * OUTC];
#pragma unroll
  for (int j = 0; j < HEADS * OUTC; ++j) {
    float a = 0.f;
#pragma unroll
    for (int d = 0; d < GO_DIM; ++d) a += h6[d] * gat_w[d * (HEADS * OUTC) + j];
    o8[j] = a;
    hn[(size_t)idx * (HEADS * OUTC) + j] = a;
  }
#pragma unroll
  for (int h = 0; h < HEADS; ++h) {
    float a = 0.f, b = 0.f;
#pragma unroll
    for (int c = 0; c < OUTC; ++c) {
      a += o8[h * OUTC + c] * gat_att[h * 2 * OUTC + c];         // att[:, :C]
      b += o8[h * OUTC + c] * gat_att[h * 2 * OUTC + OUTC + c];  // att[:, C:]
    }
    si[idx * HEADS + h] = a;
    sj[idx * HEADS + h] = b;
  }
}

// monotone float->uint key so atomicMax(uint) orders like float
__device__ __forceinline__ unsigned fkey(float f) {
  unsigned b = __float_as_uint(f);
  return b ^ (((int)b >> 31) ? 0xFFFFFFFFu : 0x80000000u);
}
__device__ __forceinline__ float funkey(unsigned k) {
  unsigned b = (k & 0x80000000u) ? (k ^ 0x80000000u) : ~k;
  return __uint_as_float(b);
}

// Stage 4a: alpha = lrelu(si[dst]+sj[src], 0.2); segment max + edge counts.
__global__ void edge_max_kernel(const int* __restrict__ ei,
                                const float* __restrict__ si,
                                const float* __restrict__ sj,
                                float* __restrict__ alpha,
                                unsigned* __restrict__ amax,
                                float* __restrict__ cnt) {
  const int idx = blockIdx.x * blockDim.x + threadIdx.x;
  if (idx >= NEDGE * BATCH * HEADS) return;
  const int e  = idx >> 5;        // 32 = B*H
  const int bh = idx & 31;        // (b*2 + h)
  const int s = ei[e], d = ei[NEDGE + e];
  float a = si[(size_t)d * 32 + bh] + sj[(size_t)s * 32 + bh];
  a = lrelu(a, 0.2f);
  alpha[idx] = a;
  atomicMax(&amax[(size_t)d * 32 + bh], fkey(a));
  if (bh == 0) atomicAdd(&cnt[d], 1.0f);
}

// Stage 4b: w = exp(alpha - amax[dst]); accumulate softmax denominator.
__global__ void edge_exp_kernel(const int* __restrict__ ei,
                                float* __restrict__ alpha,
                                const unsigned* __restrict__ amax,
                                float* __restrict__ denom) {
  const int idx = blockIdx.x * blockDim.x + threadIdx.x;
  if (idx >= NEDGE * BATCH * HEADS) return;
  const int e  = idx >> 5;
  const int bh = idx & 31;
  const int d = ei[NEDGE + e];
  const float m = funkey(amax[(size_t)d * 32 + bh]);
  const float w = __expf(alpha[idx] - m);
  alpha[idx] = w;
  atomicAdd(&denom[(size_t)d * 32 + bh], w);
}

// Stage 4c: scatter msg = hn[src] * (w / denom[dst]) into summed[dst].
__global__ void edge_scatter_kernel(const int* __restrict__ ei,
                                    const float* __restrict__ alpha,
                                    const float* __restrict__ denom,
                                    const float* __restrict__ hn,
                                    float* __restrict__ summed) {
  const int idx = blockIdx.x * blockDim.x + threadIdx.x;
  if (idx >= NEDGE * BATCH * HEADS * OUTC) return;
  const int e = idx >> 7;            // 128 = B*H*C
  const int r = idx & 127;
  const int b = r >> 3, h = (r >> 2) & 1, c = r & 3;
  const int bh = b * 2 + h;
  const int s = ei[e], d = ei[NEDGE + e];
  const float w  = alpha[(size_t)e * 32 + bh];
  const float dn = denom[(size_t)d * 32 + bh];
  const float msg = hn[((size_t)s * BATCH + b) * 8 + h * 4 + c] * (w / dn);
  atomicAdd(&summed[((size_t)d * BATCH + b) * 8 + h * 4 + c], msg);
}

// Stage 5a: aggr = mean_h(summed/max(cnt,1)) + gat_bias; o[b, c*N + n] = lrelu.
__global__ void aggr_kernel(const float* __restrict__ summed,
                            const float* __restrict__ cnt,
                            const float* __restrict__ gat_bias,
                            float* __restrict__ obuf) {
  const int idx = blockIdx.x * blockDim.x + threadIdx.x;
  if (idx >= N_GO * BATCH * OUTC) return;
  const int n = idx >> 6;            // 64 = B*C
  const int r = idx & 63;
  const int b = r >> 2, c = r & 3;
  float cc = cnt[n]; cc = cc < 1.f ? 1.f : cc;
  const size_t base = ((size_t)n * BATCH + b) * 8;
  float a = (summed[base + c] + summed[base + 4 + c]) * 0.5f / cc + gat_bias[c];
  obuf[(size_t)b * KREAD + c * N_GO + n] = lrelu(a, 0.01f);
}

// ---------------------------------------------------------------------------
// Stage 5b: readout GEMM M=16 N=32 K=16384 via WMMA bf16, K split over 32
// wave-slices writing deterministic partials.
// ---------------------------------------------------------------------------
__global__ void readout_wmma_kernel(const float* __restrict__ obuf,
                                    const float* __restrict__ out_w,
                                    float* __restrict__ rpart) {
  const int task   = blockIdx.x * 8 + (threadIdx.x >> 5);  // 0..63
  const int lane   = threadIdx.x & 31;
  const int ntile  = task & 1;        // 0..1 (16 cols each)
  const int kslice = task >> 1;       // 0..31 (512 K each)
  const int n0     = ntile * 16;
  const int mrow   = lane & 15;
  const int half   = lane >> 4;

  const float* arow = obuf  + (size_t)mrow * KREAD;
  const float* brow = out_w + (size_t)(n0 + mrow) * KREAD;

  v8f acc = {};
  const int kend = kslice * 512 + 512;
  for (int kc = kslice * 512; kc < kend; kc += 32) {
    const int k0 = kc + 8 * half;
    const int k1 = kc + 16 + 8 * half;
    v16bf A, Bm;
#pragma unroll
    for (int i = 0; i < 8; ++i) {
      A[i]      = tob(arow[k0 + i]);
      A[8 + i]  = tob(arow[k1 + i]);
      Bm[i]     = tob(brow[k0 + i]);
      Bm[8 + i] = tob(brow[k1 + i]);
    }
    acc = __builtin_amdgcn_wmma_f32_16x16x32_bf16(
        false, A, false, Bm, (short)0, acc, false, false);
  }
  const int n = n0 + mrow;
#pragma unroll
  for (int r = 0; r < 8; ++r) {
    const int b = r + 8 * half;
    rpart[(size_t)kslice * (BATCH * NUM_OUT) + b * NUM_OUT + n] = acc[r];
  }
}

__global__ void readout_final_kernel(const float* __restrict__ rpart,
                                     const float* __restrict__ out_b,
                                     float* __restrict__ out) {
  const int idx = blockIdx.x * blockDim.x + threadIdx.x;  // b*32 + r
  if (idx >= BATCH * NUM_OUT) return;
  float a = out_b[idx & (NUM_OUT - 1)];
  for (int s = 0; s < 32; ++s) a += rpart[(size_t)s * (BATCH * NUM_OUT) + idx];
  out[idx] = a;
}

// ---------------------------------------------------------------------------
extern "C" void kernel_launch(void* const* d_in, const int* in_sizes, int n_in,
                              void* d_out, int out_size, void* d_ws, size_t ws_size,
                              hipStream_t stream) {
  const float* x        = (const float*)d_in[0];
  const int*   edge     = (const int*)  d_in[1];
  const float* W_sub    = (const float*)d_in[2];
  const float* b_sub    = (const float*)d_in[3];
  const float* fc1_w    = (const float*)d_in[4];
  const float* mask_rep = (const float*)d_in[5];
  const float* fc1_b    = (const float*)d_in[6];
  const float* gat_w    = (const float*)d_in[7];
  const float* gat_att  = (const float*)d_in[8];
  const float* gat_bias = (const float*)d_in[9];
  const float* out_w    = (const float*)d_in[10];
  const float* out_b    = (const float*)d_in[11];
  float* out = (float*)d_out;

  // workspace layout (floats)
  float*    ws      = (float*)d_ws;
  float*    x_cat   = ws;                                  // 16384
  float*    h_nbd   = x_cat + 16384;                       // 393216
  float*    hn      = h_nbd + (size_t)N_GO * BATCH * GO_DIM;         // 524288
  float*    si      = hn + (size_t)N_GO * BATCH * HEADS * OUTC;      // 131072
  float*    sj      = si + (size_t)N_GO * BATCH * HEADS;             // 131072
  float*    alpha   = sj + (size_t)N_GO * BATCH * HEADS;             // 2097152
  unsigned* amax    = (unsigned*)(alpha + (size_t)NEDGE * BATCH * HEADS);
  float*    denom   = (float*)(amax + (size_t)N_GO * BATCH * HEADS); // 131072
  float*    summed  = denom + (size_t)N_GO * BATCH * HEADS;          // 524288
  float*    cnt     = summed + (size_t)N_GO * BATCH * HEADS * OUTC;  // 4096
  float*    obuf    = cnt + N_GO;                                    // 262144
  float*    rpart   = obuf + (size_t)BATCH * KREAD;                  // 16384

  // zero atomic accumulators (amax | denom | summed | cnt), contiguous
  const size_t zero_floats = (size_t)N_GO * BATCH * HEADS        // amax
                           + (size_t)N_GO * BATCH * HEADS        // denom
                           + (size_t)N_GO * BATCH * HEADS * OUTC // summed
                           + N_GO;                               // cnt
  (void)hipMemsetAsync((void*)amax, 0, zero_floats * sizeof(float), stream);

  subnet_kernel<<<G_GENES, 256, 0, stream>>>(x, W_sub, b_sub, x_cat);
  fc1_wmma_kernel<<<NFC1 / 16 / 8, 256, 0, stream>>>(x_cat, fc1_w, mask_rep, fc1_b, h_nbd);
  gat_proj_kernel<<<(N_GO * BATCH) / 256, 256, 0, stream>>>(h_nbd, gat_w, gat_att, hn, si, sj);
  edge_max_kernel<<<(NEDGE * BATCH * HEADS) / 256, 256, 0, stream>>>(edge, si, sj, alpha, amax, cnt);
  edge_exp_kernel<<<(NEDGE * BATCH * HEADS) / 256, 256, 0, stream>>>(edge, alpha, amax, denom);
  edge_scatter_kernel<<<(NEDGE * BATCH * HEADS * OUTC) / 256, 256, 0, stream>>>(edge, alpha, denom, hn, summed);
  aggr_kernel<<<(N_GO * BATCH * OUTC) / 256, 256, 0, stream>>>(summed, cnt, gat_bias, obuf);
  readout_wmma_kernel<<<8, 256, 0, stream>>>(obuf, out_w, rpart);
  readout_final_kernel<<<2, 256, 0, stream>>>(rpart, out_b, out);
}
